// Commit2Seq_35012573397405
// MI455X (gfx1250) — matrix operand order repaired and checked by reading, not compiled, not run on hardware
//
#include <hip/hip_runtime.h>
#include <math.h>

// ---------------- problem constants (from reference setup_inputs) -------------
#define BB   128          // batch
#define KK   220          // encoder length
#define HH   512          // hidden
#define VV   32000        // vocab
#define TT   32           // target_max_length
#define H3   (3*HH)       // 1536
#define XD   (2*HH)       // 1024, concat(h_new, ct)
#define KBLK (XD/32)      // 32 WMMA K-steps
#define NEG_INF (-1e30f)
#define BOS  1

typedef __attribute__((ext_vector_type(16))) _Float16 v16h;
typedef __attribute__((ext_vector_type(8)))  _Float16 v8h;
typedef __attribute__((ext_vector_type(8)))  float    v8f;

// =============================================================================
// Kernel 1: pack out_W (2H x V, f32 row-major) into f16 WMMA B-fragment order.
// Fragment layout (16x16x32 f16 B, K x N = 32 x 16):
//   lanes 0-15 : N = lane, halves p<8 hold K = p, p>=8 hold K = 16+(p-8)
//   lanes 16-31: N = lane-16, halves p<8 hold K = 8+p, p>=8 hold K = 24+(p-8)
// Storage: Wp[ ((vt*KBLK + kb)*32 + lane)*16 + p ]
// =============================================================================
__global__ void pack_w_kernel(const float* __restrict__ outW,
                              _Float16* __restrict__ Wp) {
  int idx = blockIdx.x * 256 + threadIdx.x;       // < XD*VV = 32,768,000
  int k = idx / VV;
  int v = idx - k * VV;
  int vt = v >> 4, n = v & 15;
  int kb = k >> 5, kr = k & 31;
  int lane, p;
  if (kr < 16) { lane = n + ((kr >> 3) << 4);        p = kr & 7;       }
  else         { lane = n + (((kr - 16) >> 3) << 4); p = 8 + (kr & 7); }
  size_t dst = ((((size_t)vt * KBLK + kb) * 32 + lane) << 4) + p;
  Wp[dst] = (_Float16)outW[idx];                  // coalesced read, 32B-granule scatter store
}

// ------------------------- block reductions (256 thr) ------------------------
__device__ __forceinline__ float blockReduceMax(float v, float* red, int tid) {
  red[tid] = v; __syncthreads();
  for (int s = 128; s > 0; s >>= 1) {
    if (tid < s) red[tid] = fmaxf(red[tid], red[tid + s]);
    __syncthreads();
  }
  float r = red[0]; __syncthreads();
  return r;
}
__device__ __forceinline__ float blockReduceSum(float v, float* red, int tid) {
  red[tid] = v; __syncthreads();
  for (int s = 128; s > 0; s >>= 1) {
    if (tid < s) red[tid] += red[tid + s];
    __syncthreads();
  }
  float r = red[0]; __syncthreads();
  return r;
}

// ------------------------- masked attention helper ---------------------------
// wh = W h ; s[k] = E[b,k,:].wh (masked) ; a = softmax(s) ; ct = sum_k a[k] E[b,k,:]
__device__ void attend(const float* __restrict__ E, const float* __restrict__ W,
                       int len, int b, const float* sh_h,
                       float* sh_wh, float* sh_s, float* sh_red,
                       float* sh_ct, int tid) {
  for (int r = tid; r < HH; r += 256) {           // wh[r] = W[r,:] . h
    const float* wr = W + (size_t)r * HH;
    float acc = 0.f;
    #pragma unroll 4
    for (int j = 0; j < HH; ++j) acc += wr[j] * sh_h[j];
    sh_wh[r] = acc;
  }
  __syncthreads();
  for (int k = tid; k < KK; k += 256) {           // scores (masked)
    float s = NEG_INF;
    if (k < len) {
      const float* er = E + ((size_t)b * KK + k) * HH;
      s = 0.f;
      #pragma unroll 4
      for (int j = 0; j < HH; ++j) s += er[j] * sh_wh[j];
    }
    sh_s[k] = s;
  }
  __syncthreads();
  float lm = -INFINITY;
  for (int k = tid; k < KK; k += 256) lm = fmaxf(lm, sh_s[k]);
  float m = blockReduceMax(lm, sh_red, tid);
  float ls = 0.f;
  for (int k = tid; k < KK; k += 256) { float e = expf(sh_s[k] - m); sh_s[k] = e; ls += e; }
  float tot = blockReduceSum(ls, sh_red, tid);    // sync inside makes sh_s visible
  float inv = 1.f / tot;
  for (int r = tid; r < HH; r += 256) {           // context
    float acc = 0.f;
    for (int k = 0; k < KK; ++k) acc += sh_s[k] * E[((size_t)b * KK + k) * HH + r];
    sh_ct[r] = acc * inv;
  }
  __syncthreads();
}

// =============================================================================
// Kernel 2: per-step attention + GRU. One block per batch row, 256 threads.
// Produces h_{t+1} (f32, hbuf) and X = concat(h_new, ct) in f16 (row-major).
// =============================================================================
__global__ void __launch_bounds__(256)
attn_gru_kernel(const float* __restrict__ E_del, const float* __restrict__ E_add,
                const float* __restrict__ ehd,   const float* __restrict__ eha,
                const float* __restrict__ W_del, const float* __restrict__ W_add,
                const float* __restrict__ W3,    const float* __restrict__ emb,
                const float* __restrict__ W_ih,  const float* __restrict__ W_hh,
                const float* __restrict__ b_ih,  const float* __restrict__ b_hh,
                const int* __restrict__ len_del, const int* __restrict__ len_add,
                float* __restrict__ hbuf, const int* __restrict__ tokbuf,
                _Float16* __restrict__ xf16, int step) {
  __shared__ float sh_h[HH], sh_wh[HH], sh_ct_del[HH], sh_ct_add[HH];
  __shared__ float sh_s[KK + 4], sh_red[256];
  __shared__ float sh_x[HH], sh_gi[H3], sh_gh[H3];
  const int b = blockIdx.x, tid = threadIdx.x;

  int tok;
  if (step == 0) {
    for (int j = tid; j < HH; j += 256)
      sh_h[j] = 0.5f * (ehd[(size_t)b * HH + j] + eha[(size_t)b * HH + j]);
    tok = BOS;
  } else {
    for (int j = tid; j < HH; j += 256) sh_h[j] = hbuf[(size_t)b * HH + j];
    tok = tokbuf[b];
  }
  __syncthreads();

  attend(E_del, W_del, len_del[b], b, sh_h, sh_wh, sh_s, sh_red, sh_ct_del, tid);
  attend(E_add, W_add, len_add[b], b, sh_h, sh_wh, sh_s, sh_red, sh_ct_add, tid);

  // third attention over {ct_del, ct_add}
  for (int r = tid; r < HH; r += 256) {           // wh3 into sh_wh
    const float* wr = W3 + (size_t)r * HH;
    float acc = 0.f;
    #pragma unroll 4
    for (int j = 0; j < HH; ++j) acc += wr[j] * sh_h[j];
    sh_wh[r] = acc;
  }
  __syncthreads();
  float p0 = 0.f, p1 = 0.f;
  for (int r = tid; r < HH; r += 256) { p0 += sh_ct_del[r] * sh_wh[r]; p1 += sh_ct_add[r] * sh_wh[r]; }
  float s0 = blockReduceSum(p0, sh_red, tid);
  float s1 = blockReduceSum(p1, sh_red, tid);
  float mm = fmaxf(s0, s1);
  float e0 = expf(s0 - mm), e1 = expf(s1 - mm);
  float a0 = e0 / (e0 + e1), a1 = e1 / (e0 + e1);
  for (int r = tid; r < HH; r += 256)             // final ct overwrites sh_wh
    sh_wh[r] = a0 * sh_ct_del[r] + a1 * sh_ct_add[r];

  // GRU cell
  for (int i = tid; i < HH; i += 256) sh_x[i] = emb[(size_t)tok * HH + i];
  __syncthreads();
  for (int j = tid; j < H3; j += 256) {
    float gi = b_ih[j], gh = b_hh[j];
    #pragma unroll 4
    for (int i = 0; i < HH; ++i) {
      gi += sh_x[i] * W_ih[(size_t)i * H3 + j];   // coalesced across threads
      gh += sh_h[i] * W_hh[(size_t)i * H3 + j];
    }
    sh_gi[j] = gi; sh_gh[j] = gh;
  }
  __syncthreads();
  for (int j = tid; j < HH; j += 256) {
    float r = 1.f / (1.f + expf(-(sh_gi[j]        + sh_gh[j])));
    float z = 1.f / (1.f + expf(-(sh_gi[HH + j]   + sh_gh[HH + j])));
    float n = tanhf(sh_gi[2*HH + j] + r * sh_gh[2*HH + j]);
    float hn = (1.f - z) * n + z * sh_h[j];
    hbuf[(size_t)b * HH + j] = hn;
    xf16[(size_t)b * XD + j]      = (_Float16)hn;
    xf16[(size_t)b * XD + HH + j] = (_Float16)sh_wh[j];
  }
}

// =============================================================================
// Kernel 3: logits = X (B x 1024, f16) @ Wp (packed f16) + bias, via
// v_wmma_f32_16x16x32_f16. 8 waves/block -> 16(M) x 128(N) tile.
// grid = (V/128, B/16). Wp is L2-resident (65.5 MB < 192 MB).
// =============================================================================
__global__ void __launch_bounds__(256)
gemm_wmma_kernel(const _Float16* __restrict__ Wp, const _Float16* __restrict__ X,
                 const float* __restrict__ bias, float* __restrict__ out) {
  const int wave = threadIdx.x >> 5;
  const int lane = threadIdx.x & 31;
  const int vt = blockIdx.x * 8 + wave;           // 16-col tile index, 0..1999
  const int mb = blockIdx.y;                      // 16-row tile index, 0..7

  v8f c = {};
  // A fragment addressing (ISA 7.12.2, 16-bit A 16x32):
  //   M = lane&15 ; halves p<8 -> K = base+p, p>=8 -> K = base+16+(p-8); lanes>=16: base += 8
  const _Float16* xrow  = X + (size_t)(mb * 16 + (lane & 15)) * XD;
  const int kofsA = (lane >= 16) ? 8 : 0;
  const _Float16* wbase = Wp + (((size_t)vt * KBLK * 32 + lane) << 4);

  for (int kb = 0; kb < KBLK; ++kb) {
    int k0 = kb * 32 + kofsA;
    v8h c0 = *(const v8h*)(xrow + k0);            // K = k0 .. k0+7
    v8h c1 = *(const v8h*)(xrow + k0 + 16);       // K = k0+16 .. k0+23
    union { v16h v; v8h h[2]; } ua; ua.h[0] = c0; ua.h[1] = c1;
    v16h bfrag = *(const v16h*)(wbase + ((size_t)kb << 9));  // 16 halves, coalesced
    c = __builtin_amdgcn_wmma_f32_16x16x32_f16(false, ua.v, false, bfrag,
                                               (short)0, c, false, false);
  }

  // C layout: VGPR r -> M = r (lanes 0-15) / M = 8+r (lanes 16-31), N = lane&15
  const int n  = (vt << 4) + (lane & 15);
  const float bv = bias[n];
  const int mrow = mb * 16 + ((lane >= 16) ? 8 : 0);
  #pragma unroll
  for (int r = 0; r < 8; ++r)
    out[(size_t)(mrow + r) * VV + n] = c[r] + bv;
}

// =============================================================================
// Kernel 4: in-place log_softmax over V per row + greedy argmax (first-max tie).
// =============================================================================
__global__ void __launch_bounds__(256)
logsoftmax_argmax_kernel(float* __restrict__ logits, int* __restrict__ tok) {
  __shared__ float redv[256];
  __shared__ int   redi[256];
  const int b = blockIdx.x, tid = threadIdx.x;
  float* row = logits + (size_t)b * VV;

  float lm = -INFINITY; int li = 0;
  for (int v = tid; v < VV; v += 256) {
    float x = row[v];
    if (x > lm) { lm = x; li = v; }               // ascending scan -> first max
  }
  redv[tid] = lm; redi[tid] = li; __syncthreads();
  for (int s = 128; s > 0; s >>= 1) {
    if (tid < s) {
      float o = redv[tid + s]; int oi = redi[tid + s];
      if (o > redv[tid] || (o == redv[tid] && oi < redi[tid])) { redv[tid] = o; redi[tid] = oi; }
    }
    __syncthreads();
  }
  const float m = redv[0]; const int am = redi[0]; __syncthreads();

  float ls = 0.f;
  for (int v = tid; v < VV; v += 256) ls += expf(row[v] - m);
  redv[tid] = ls; __syncthreads();
  for (int s = 128; s > 0; s >>= 1) { if (tid < s) redv[tid] += redv[tid + s]; __syncthreads(); }
  const float lse = m + logf(redv[0]);

  for (int v = tid; v < VV; v += 256) row[v] = row[v] - lse;
  if (tid == 0) tok[b] = am;
}

// =============================================================================
extern "C" void kernel_launch(void* const* d_in, const int* in_sizes, int n_in,
                              void* d_out, int out_size, void* d_ws, size_t ws_size,
                              hipStream_t stream) {
  (void)in_sizes; (void)n_in; (void)out_size; (void)ws_size;
  const float* E_del  = (const float*)d_in[0];
  const float* E_add  = (const float*)d_in[1];
  const float* ehd    = (const float*)d_in[2];
  const float* eha    = (const float*)d_in[3];
  const float* W_del  = (const float*)d_in[4];
  const float* W_add  = (const float*)d_in[5];
  const float* W3     = (const float*)d_in[6];
  const float* emb    = (const float*)d_in[7];
  const float* W_ih   = (const float*)d_in[8];
  const float* W_hh   = (const float*)d_in[9];
  const float* b_ih   = (const float*)d_in[10];
  const float* b_hh   = (const float*)d_in[11];
  const float* outW   = (const float*)d_in[12];
  const float* outb   = (const float*)d_in[13];
  const int* len_del  = (const int*)d_in[14];
  const int* len_add  = (const int*)d_in[15];
  float* out = (float*)d_out;

  // workspace layout
  char* ws = (char*)d_ws;
  _Float16* Wp   = (_Float16*)ws;                         // XD*VV*2      = 65,536,000 B
  size_t off = (size_t)XD * VV * sizeof(_Float16);
  float* hbuf    = (float*)(ws + off);  off += (size_t)BB * HH * sizeof(float);   // 262,144 B
  _Float16* xf16 = (_Float16*)(ws + off); off += (size_t)BB * XD * sizeof(_Float16); // 262,144 B
  int* tokbuf    = (int*)(ws + off);

  // one-time per call: pack out_W into f16 WMMA fragments (L2-resident afterwards)
  pack_w_kernel<<<(XD * VV) / 256, 256, 0, stream>>>(outW, Wp);

  for (int t = 0; t < TT; ++t) {
    attn_gru_kernel<<<BB, 256, 0, stream>>>(E_del, E_add, ehd, eha, W_del, W_add, W3,
                                            emb, W_ih, W_hh, b_ih, b_hh,
                                            len_del, len_add, hbuf, tokbuf, xf16, t);
    float* logits = out + (size_t)t * BB * VV;
    gemm_wmma_kernel<<<dim3(VV / 128, BB / 16), 256, 0, stream>>>(Wp, xf16, outb, logits);
    logsoftmax_argmax_kernel<<<BB, 256, 0, stream>>>(logits, tokbuf);
  }
}
